// CausalSelfAttention_56049323213685
// MI455X (gfx1250) — matrix operand order repaired and checked by reading, not compiled, not run on hardware
//
#include <hip/hip_runtime.h>

// ---------------------------------------------------------------------------
// CDNA5 (gfx1250, wave32) causal self-attention:
//   qkv = x @ w_qkv^T  -> RoPE(q,k) -> flash attention -> out = attn @ w_out^T
// Matmuls: v_wmma_f32_16x16x32_bf16 (fp32 accumulate).
// K/V tile staging: global_load_async_to_lds_b128 + s_wait_asynccnt
// (double-buffered). Softmax row reductions: ds_swizzle_b32 XOR butterflies.
// ---------------------------------------------------------------------------

#define B_   2
#define S_   2048
#define D_   2048
#define H_   16
#define HD_  128
#define N3_  6144          // 3*D
#define M_   4096          // B*S

#define AS1 __attribute__((address_space(1)))
#define AS3 __attribute__((address_space(3)))

#if defined(__has_builtin)
#if __has_builtin(__builtin_amdgcn_global_load_async_to_lds_b128) && \
    __has_builtin(__builtin_amdgcn_s_wait_asynccnt)
#define USE_ASYNC_LDS 1
#endif
#if __has_builtin(__builtin_amdgcn_ds_swizzle)
#define USE_DS_SWIZZLE 1
#endif
#endif

typedef __attribute__((ext_vector_type(16))) __bf16 v16bf;
typedef __attribute__((ext_vector_type(8)))  float  v8f;
typedef __attribute__((ext_vector_type(4)))  int    v4i;

union Frag {
  v16bf v;
  unsigned int u[8];
};

__device__ __forceinline__ unsigned short f32_to_bf16(float f) {
  unsigned int u = __float_as_uint(f);
  unsigned int r = u + 0x7FFFu + ((u >> 16) & 1u);   // round-to-nearest-even
  return (unsigned short)(r >> 16);
}
__device__ __forceinline__ float bf16_to_f32(unsigned short h) {
  return __uint_as_float(((unsigned int)h) << 16);
}

__device__ __forceinline__ v8f wmma_bf16(const Frag& a, const Frag& b, v8f c) {
  return __builtin_amdgcn_wmma_f32_16x16x32_bf16(
      /*neg_a=*/false, a.v, /*neg_b=*/false, b.v,
      /*c_mod=*/(short)0, c, /*reuse_a=*/false, /*reuse_b=*/false);
}

__device__ __forceinline__ v8f v8f_zero() {
  v8f z;
  for (int i = 0; i < 8; ++i) z[i] = 0.0f;
  return z;
}

// Row reductions across the 16 lanes of a half-wave. XOR masks 1,2,4,8 never
// cross the 16-lane boundary, so a full-wave ds_swizzle XOR works per half.
#ifdef USE_DS_SWIZZLE
#define SWZ_XOR(x, imm) \
  __int_as_float(__builtin_amdgcn_ds_swizzle(__float_as_int(x), imm))
__device__ __forceinline__ float rowmax16(float x) {
  x = fmaxf(x, SWZ_XOR(x, 0x041f));   // xor 1
  x = fmaxf(x, SWZ_XOR(x, 0x081f));   // xor 2
  x = fmaxf(x, SWZ_XOR(x, 0x101f));   // xor 4
  x = fmaxf(x, SWZ_XOR(x, 0x201f));   // xor 8
  return x;
}
__device__ __forceinline__ float rowsum16(float x) {
  x += SWZ_XOR(x, 0x041f);
  x += SWZ_XOR(x, 0x081f);
  x += SWZ_XOR(x, 0x101f);
  x += SWZ_XOR(x, 0x201f);
  return x;
}
#else
__device__ __forceinline__ float rowmax16(float x) {
  for (int off = 1; off < 16; off <<= 1) x = fmaxf(x, __shfl_xor(x, off, 16));
  return x;
}
__device__ __forceinline__ float rowsum16(float x) {
  for (int off = 1; off < 16; off <<= 1) x += __shfl_xor(x, off, 16);
  return x;
}
#endif

// ---------------------------------------------------------------------------
// GEMM: C[m,n] = sum_k A[m,k] * W[n,k]    (W fp32 row-major NxK)
// A is fp32 (A_BF16=false) or bf16 (A_BF16=true). Output fp32 or bf16.
// Block: 64x64 tile, 128 threads = 4 waves; wave w covers rows w*16..w*16+15.
// ---------------------------------------------------------------------------
template <bool A_BF16, bool OUT_F32>
__global__ __launch_bounds__(128)
void gemm_bf16_kernel(const float* __restrict__ Af32,
                      const unsigned short* __restrict__ Abf,
                      const float* __restrict__ W,
                      float* __restrict__ outF,
                      unsigned short* __restrict__ outB,
                      int M, int N, int K) {
  __shared__ unsigned short As[64][34];
  __shared__ unsigned short Bs[64][34];

  const int nBase = blockIdx.x * 64;
  const int mBase = blockIdx.y * 64;
  const int tid   = threadIdx.x;
  const int wave  = tid >> 5;
  const int lane  = tid & 31;
  const int l16   = lane & 15;
  const int hgrp  = lane >> 4;        // lane half (0/1)

  v8f acc[4];
  for (int i = 0; i < 4; ++i) acc[i] = v8f_zero();

  for (int kB = 0; kB < K; kB += 32) {
    __syncthreads();
    // stage 64x32 A tile and 64x32 W tile, converting to bf16
    for (int i = 0; i < 16; ++i) {
      int idx = tid + i * 128;          // 0..2047
      int r = idx >> 5, c = idx & 31;
      float av;
      if (A_BF16)
        av = bf16_to_f32(Abf[(size_t)(mBase + r) * K + kB + c]);
      else
        av = Af32[(size_t)(mBase + r) * K + kB + c];
      As[r][c] = f32_to_bf16(av);
      Bs[r][c] = f32_to_bf16(W[(size_t)(nBase + r) * K + kB + c]);
    }
    // prefetch next K-slab (lowers to global_prefetch_b8)
    if (kB + 32 < K) {
      int r = tid >> 1, c = (tid & 1) * 16;
      __builtin_prefetch(&W[(size_t)(nBase + r) * K + kB + 32 + c], 0, 0);
      if (!A_BF16)
        __builtin_prefetch(&Af32[(size_t)(mBase + r) * K + kB + 32 + c], 0, 0);
      else
        __builtin_prefetch(&Abf[(size_t)(mBase + r) * K + kB + 32 + c], 0, 0);
    }
    __syncthreads();

    // A fragment: row = wave*16 + l16, K per ISA A-layout
    Frag afr;
    int arow = wave * 16 + l16;
    #pragma unroll
    for (int v = 0; v < 8; ++v) {
      int k0 = (v < 4) ? (hgrp * 8 + v * 2) : (16 + hgrp * 8 + (v - 4) * 2);
      afr.u[v] = *(const unsigned int*)&As[arow][k0];
    }
    #pragma unroll
    for (int ct = 0; ct < 4; ++ct) {
      Frag bfr;
      int brow = ct * 16 + l16;         // output column (N)
      #pragma unroll
      for (int v = 0; v < 8; ++v) {
        int k0 = (v < 4) ? (hgrp * 8 + v * 2) : (16 + hgrp * 8 + (v - 4) * 2);
        bfr.u[v] = *(const unsigned int*)&Bs[brow][k0];
      }
      acc[ct] = wmma_bf16(afr, bfr, acc[ct]);
    }
  }

  // epilogue: C layout row = v + 8*hgrp, col = l16
  #pragma unroll
  for (int ct = 0; ct < 4; ++ct) {
    #pragma unroll
    for (int v = 0; v < 8; ++v) {
      int m = mBase + wave * 16 + v + 8 * hgrp;
      int n = nBase + ct * 16 + l16;
      float val = acc[ct][v];
      if (OUT_F32) outF[(size_t)m * N + n] = val;
      else         outB[(size_t)m * N + n] = f32_to_bf16(val);
    }
  }
}

// ---------------------------------------------------------------------------
// RoPE in-place on bf16 qkv[B,S,3D]; folds 1/sqrt(HD) into q.
// One thread per (b,s,h,j<64).
// ---------------------------------------------------------------------------
__global__ __launch_bounds__(256)
void rope_kernel(unsigned short* __restrict__ qkv,
                 const int* __restrict__ start_pos_p) {
  int idx = blockIdx.x * blockDim.x + threadIdx.x;   // < B*S*H*64
  int j  = idx & 63;
  int h  = (idx >> 6) & 15;
  int bs = idx >> 10;                                // b*S + s
  int s  = bs & (S_ - 1);

  float pos = (float)(s + start_pos_p[0]);
  float inv = __expf(-((float)j * (1.0f / 64.0f)) * logf(10000.0f));
  float ang = pos * inv;
  float cs = __cosf(ang), sn = __sinf(ang);

  size_t ro = (size_t)bs * N3_ + (size_t)h * HD_ + j;
  const float sc = 0.08838834764831845f;             // 1/sqrt(128)

  float q1 = bf16_to_f32(qkv[ro]);
  float q2 = bf16_to_f32(qkv[ro + 64]);
  qkv[ro]      = f32_to_bf16((q1 * cs - q2 * sn) * sc);
  qkv[ro + 64] = f32_to_bf16((q2 * cs + q1 * sn) * sc);

  size_t ko = ro + D_;
  float k1 = bf16_to_f32(qkv[ko]);
  float k2 = bf16_to_f32(qkv[ko + 64]);
  qkv[ko]      = f32_to_bf16(k1 * cs - k2 * sn);
  qkv[ko + 64] = f32_to_bf16(k2 * cs + k1 * sn);
}

// ---------------------------------------------------------------------------
// Flash attention. Grid (B*H, S/64); 128 threads = 4 waves.
// Double-buffered K/V tiles staged via async global->LDS copies.
// Per 32-key block: 8 WMMA for scores, 8 WMMA for PV.
// ---------------------------------------------------------------------------
__global__ __launch_bounds__(128)
void flash_attn_kernel(const unsigned short* __restrict__ qkv,
                       unsigned short* __restrict__ attn_out) {
  __shared__ unsigned short Ks[2][32][136];   // 272B rows: 16B-aligned for b128
  __shared__ unsigned short Vs[2][32][136];
  __shared__ unsigned short Ps[4][16][34];

  const int bh = blockIdx.x;
  const int b  = bh >> 4;
  const int h  = bh & 15;
  const int qtile = blockIdx.y;
  const int tid  = threadIdx.x;
  const int wave = tid >> 5;
  const int lane = tid & 31;
  const int l16  = lane & 15;
  const int hgrp = lane >> 4;

  const int qb = qtile * 64 + wave * 16;
  const size_t base = (size_t)b * S_ * N3_;

  // Q fragments for this wave (A-layout, K-dim = head dim, 4 chunks of 32)
  Frag qf[4];
  {
    const unsigned short* qrow = qkv + base + (size_t)(qb + l16) * N3_ + h * HD_;
    #pragma unroll
    for (int c = 0; c < 4; ++c)
      #pragma unroll
      for (int v = 0; v < 8; ++v) {
        int hd = c * 32 + ((v < 4) ? (hgrp * 8 + v * 2)
                                   : (16 + hgrp * 8 + (v - 4) * 2));
        qf[c].u[v] = *(const unsigned int*)(qrow + hd);
      }
  }

  v8f O[8];
  float mrow[8], lsum[8];
  #pragma unroll
  for (int i = 0; i < 8; ++i) { O[i] = v8f_zero(); mrow[i] = -1e30f; lsum[i] = 0.0f; }

  // tile stager: copies 32x128 bf16 K and V tiles for key block tb into buffer
  auto stage = [&](int tb, int bufsel) {
#ifdef USE_ASYNC_LDS
    #pragma unroll
    for (int i = 0; i < 4; ++i) {
      int idx = tid + i * 128;              // 0..511, 8 ushorts each
      int key = idx >> 4;
      int hd  = (idx & 15) * 8;
      const unsigned short* src =
          qkv + base + (size_t)(tb + key) * N3_ + D_ + h * HD_ + hd;
      __builtin_amdgcn_global_load_async_to_lds_b128(
          (AS1 v4i*)src, (AS3 v4i*)&Ks[bufsel][key][hd], 0, 0);
      __builtin_amdgcn_global_load_async_to_lds_b128(
          (AS1 v4i*)(src + D_), (AS3 v4i*)&Vs[bufsel][key][hd], 0, 0);
    }
#else
    for (int i = 0; i < 32; ++i) {
      int idx = tid + i * 128;              // 0..4095
      int key = idx >> 7, hd = idx & 127;
      const unsigned short* src =
          qkv + base + (size_t)(tb + key) * N3_ + D_ + h * HD_ + hd;
      Ks[bufsel][key][hd] = *src;
      Vs[bufsel][key][hd] = *(src + D_);
    }
#endif
  };

  const int nblk = qtile * 2 + 2;            // causal: keys up to qtile*64+63
  stage(0, 0);
  for (int blk = 0; blk < nblk; ++blk) {
    const int tb  = blk * 32;
    const int cur = blk & 1;
    if (blk + 1 < nblk) {
      stage((blk + 1) * 32, cur ^ 1);        // prefetch next tile (other buffer)
#ifdef USE_ASYNC_LDS
      __builtin_amdgcn_s_wait_asynccnt(8);   // 8 newest in flight => cur landed
#endif
    } else {
#ifdef USE_ASYNC_LDS
      __builtin_amdgcn_s_wait_asynccnt(0);
#endif
    }
    __syncthreads();

    if (tb <= qb + 15) {                     // wave-uniform: any unmasked key?
      v8f c0 = v8f_zero(), c1 = v8f_zero();
      #pragma unroll
      for (int ch = 0; ch < 4; ++ch) {
        Frag bk0, bk1;
        #pragma unroll
        for (int v = 0; v < 8; ++v) {
          int hd = ch * 32 + ((v < 4) ? (hgrp * 8 + v * 2) : (16 + hgrp * 8 + (v - 4) * 2));
          bk0.u[v] = *(const unsigned int*)&Ks[cur][l16][hd];
          bk1.u[v] = *(const unsigned int*)&Ks[cur][16 + l16][hd];
        }
        c0 = wmma_bf16(qf[ch], bk0, c0);
        c1 = wmma_bf16(qf[ch], bk1, c1);
      }

      // causal mask + online softmax (rows = v + 8*hgrp of this wave)
      #pragma unroll
      for (int v = 0; v < 8; ++v) {
        int q   = qb + v + 8 * hgrp;
        float s0 = ((tb + l16)      <= q) ? c0[v] : -1e30f;
        float s1 = ((tb + 16 + l16) <= q) ? c1[v] : -1e30f;
        float mx = rowmax16(fmaxf(s0, s1));
        float nm    = fmaxf(mrow[v], mx);
        float scale = __expf(mrow[v] - nm);
        mrow[v] = nm;
        float p0 = __expf(s0 - nm);
        float p1 = __expf(s1 - nm);
        float rs = rowsum16(p0 + p1);
        lsum[v] = lsum[v] * scale + rs;
        #pragma unroll
        for (int ct = 0; ct < 8; ++ct) O[ct][v] *= scale;
        // store P tile (row = v + 8*hgrp, key = l16 / 16+l16) as bf16
        Ps[wave][v + 8 * hgrp][l16]      = f32_to_bf16(p0);
        Ps[wave][v + 8 * hgrp][16 + l16] = f32_to_bf16(p1);
      }

      // reload P as A-fragment (16 rows x 32 keys)
      Frag pf;
      #pragma unroll
      for (int v = 0; v < 8; ++v) {
        int k0 = (v < 4) ? (hgrp * 8 + v * 2) : (16 + hgrp * 8 + (v - 4) * 2);
        pf.u[v] = *(const unsigned int*)&Ps[wave][l16][k0];
      }
      // PV accumulate over 8 head-dim column tiles
      #pragma unroll
      for (int ct = 0; ct < 8; ++ct) {
        Frag vf;
        #pragma unroll
        for (int v = 0; v < 8; ++v) {
          int k0 = (v < 4) ? (hgrp * 8 + v * 2) : (16 + hgrp * 8 + (v - 4) * 2);
          int hd = ct * 16 + l16;
          unsigned int lo = Vs[cur][k0][hd];
          unsigned int hi = Vs[cur][k0 + 1][hd];
          vf.u[v] = lo | (hi << 16);
        }
        O[ct] = wmma_bf16(pf, vf, O[ct]);
      }
    }
    __syncthreads();                         // all waves done with 'cur' buffer
  }

  // normalize and write attn_out bf16 [B,S,D]
  #pragma unroll
  for (int ct = 0; ct < 8; ++ct) {
    #pragma unroll
    for (int v = 0; v < 8; ++v) {
      int srow = qb + v + 8 * hgrp;
      int hd   = ct * 16 + l16;
      float o  = O[ct][v] / lsum[v];
      attn_out[(size_t)(b * S_ + srow) * D_ + h * HD_ + hd] = f32_to_bf16(o);
    }
  }
}

// ---------------------------------------------------------------------------
extern "C" void kernel_launch(void* const* d_in, const int* in_sizes, int n_in,
                              void* d_out, int out_size, void* d_ws, size_t ws_size,
                              hipStream_t stream) {
  const float* x      = (const float*)d_in[0];
  const float* w_qkv  = (const float*)d_in[1];
  const float* w_out  = (const float*)d_in[2];
  const int*   start_pos = (const int*)d_in[4];   // d_in[3] = layer_idx (unused)

  unsigned short* qkv_ws  = (unsigned short*)d_ws;                 // [M, 3D] bf16
  unsigned short* attn_ws = qkv_ws + (size_t)M_ * N3_;             // [M, D] bf16
  float* out = (float*)d_out;

  // 1) qkv = x @ w_qkv^T  (fp32 in, bf16 out)
  gemm_bf16_kernel<false, false><<<dim3(N3_ / 64, M_ / 64), 128, 0, stream>>>(
      x, nullptr, w_qkv, nullptr, qkv_ws, M_, N3_, D_);

  // 2) RoPE on q,k (in place), fold 1/sqrt(HD) into q
  rope_kernel<<<(B_ * S_ * H_ * 64) / 256, 256, 0, stream>>>(qkv_ws, start_pos);

  // 3) causal flash attention -> attn_ws bf16 [B,S,D]
  flash_attn_kernel<<<dim3(B_ * H_, S_ / 64), 128, 0, stream>>>(qkv_ws, attn_ws);

  // 4) out = attn @ w_out^T  (bf16 in, fp32 out)
  gemm_bf16_kernel<true, true><<<dim3(D_ / 64, M_ / 64), 128, 0, stream>>>(
      nullptr, attn_ws, w_out, out, nullptr, M_, D_, D_);
}
